// MoEMLP_20641612825251
// MI455X (gfx1250) — matrix operand order repaired and checked by reading, not compiled, not run on hardware
//
#include <hip/hip_runtime.h>
#include <hip/hip_bf16.h>
#include <stdint.h>

#define D_DIM 1024
#define F_DIM 2816
#define NEXP 8
#define NTOK 4096
#define MT 64      // token-tile rows
#define NT 128     // col tile

typedef __attribute__((ext_vector_type(16))) __bf16 bf16x16;
typedef __attribute__((ext_vector_type(8)))  float  f32x8;
typedef __attribute__((ext_vector_type(4)))  float  f32x4;
typedef __attribute__((ext_vector_type(4)))  unsigned int u32x4;

union FragB16 { u32x4 q[2]; bf16x16 v; };

__device__ __forceinline__ void async_copy_b128(uint32_t lds_off, uint64_t gaddr) {
    asm volatile("global_load_async_to_lds_b128 %0, %1, off"
                 :: "v"(lds_off), "v"(gaddr) : "memory");
}
__device__ __forceinline__ void wait_async() {
    asm volatile("s_wait_asynccnt 0x0" ::: "memory");
}
__device__ __forceinline__ uint32_t pack2bf16(float lo, float hi) {
    union { __bf16 h; unsigned short s; } a, b;
    a.h = (__bf16)lo; b.h = (__bf16)hi;
    return (uint32_t)a.s | ((uint32_t)b.s << 16);
}

// ---------------- init: zero output accumulation region + expert counters ----
__global__ __launch_bounds__(256) void init_kernel(float* __restrict__ out,
                                                   int* __restrict__ counts) {
    size_t i = (size_t)blockIdx.x * 256 + threadIdx.x;   // 4096 blocks -> 1M f32x4
    ((f32x4*)out)[i] = (f32x4){0.f, 0.f, 0.f, 0.f};
    if (blockIdx.x == 0 && threadIdx.x < NEXP) counts[threadIdx.x] = 0;
}

// ---------------- fp32 -> bf16 activation convert ---------------------------
__global__ __launch_bounds__(256) void cvt_kernel(const float* __restrict__ x,
                                                  __bf16* __restrict__ xbf) {
    size_t i = ((size_t)blockIdx.x * 256 + threadIdx.x) * 4;
    f32x4 v = *(const f32x4*)(x + i);
    uint32_t p0 = pack2bf16(v.x, v.y);
    uint32_t p1 = pack2bf16(v.z, v.w);
    *(uint32_t*)(xbf + i)     = p0;
    *(uint32_t*)(xbf + i + 2) = p1;
}

// ---------------- router: logits + softmax + top2 + compaction --------------
__global__ __launch_bounds__(256) void router_kernel(
    const float* __restrict__ x, const float* __restrict__ gate_w,
    float* __restrict__ logits_out, int* __restrict__ counts,
    int* __restrict__ lists, float* __restrict__ wts) {
    const int lane = threadIdx.x & 31;
    const int wave = threadIdx.x >> 5;
    const int t = blockIdx.x * 8 + wave;          // 512 blocks * 8 waves = 4096 tokens
    const float* xr = x + (size_t)t * D_DIM;
    float acc[NEXP] = {};
    for (int d = lane; d < D_DIM; d += 32) {
        float xv = xr[d];
        #pragma unroll
        for (int e = 0; e < NEXP; ++e) acc[e] += xv * gate_w[e * D_DIM + d];
    }
    #pragma unroll
    for (int e = 0; e < NEXP; ++e)
        #pragma unroll
        for (int off = 16; off > 0; off >>= 1)
            acc[e] += __shfl_xor(acc[e], off, 32);
    if (lane == 0) {
        float mx = acc[0];
        #pragma unroll
        for (int e = 1; e < NEXP; ++e) mx = fmaxf(mx, acc[e]);
        float p[NEXP];
        #pragma unroll
        for (int e = 0; e < NEXP; ++e) {
            logits_out[(size_t)t * NEXP + e] = acc[e];
            p[e] = __expf(acc[e] - mx);   // top-2 renorm cancels the denominator
        }
        int i0 = 0;
        #pragma unroll
        for (int e = 1; e < NEXP; ++e) if (p[e] > p[i0]) i0 = e;
        int i1 = (i0 == 0) ? 1 : 0;
        #pragma unroll
        for (int e = 0; e < NEXP; ++e) if (e != i0 && p[e] > p[i1]) i1 = e;
        float inv = 1.0f / (p[i0] + p[i1]);
        float w0 = p[i0] * inv, w1 = p[i1] * inv;
        int pos0 = atomicAdd(&counts[i0], 1);
        lists[i0 * NTOK + pos0] = t * 2 + 0; wts[i0 * NTOK + pos0] = w0;
        int pos1 = atomicAdd(&counts[i1], 1);
        lists[i1 * NTOK + pos1] = t * 2 + 1; wts[i1 * NTOK + pos1] = w1;
    }
}

// ---------------- gate/up GEMM + silu*up, bf16 WMMA -------------------------
__global__ __launch_bounds__(256) void gateup_kernel(
    const __bf16* __restrict__ xbf, const float* __restrict__ gate_ws,
    const float* __restrict__ up_ws, const int* __restrict__ counts,
    const int* __restrict__ lists, __bf16* __restrict__ hbuf) {
    const int e = blockIdx.z;
    const int cnt = counts[e];
    const int m0 = blockIdx.y * MT;
    if (m0 >= cnt) return;
    const int n0 = blockIdx.x * NT;
    const int tid = threadIdx.x;
    const int lane = tid & 31, wave = tid >> 5;
    const int wm = wave & 3, wn = wave >> 2;     // 4 m-subtiles x 2 col-groups

    __shared__ __bf16 ldsA[MT * 32];             // token rows, one K-chunk
    __shared__ __bf16 ldsG[NT * 40];             // transposed bf16 B (gate)
    __shared__ __bf16 ldsU[NT * 40];             // transposed bf16 B (up)
    __shared__ int sEnt[MT];
    __shared__ int sRow[MT];

    if (tid < MT) {
        int idx = m0 + tid; if (idx >= cnt) idx = cnt - 1;
        int ent = lists[e * NTOK + idx];
        sEnt[tid] = ent; sRow[tid] = ent >> 1;
    }
    __syncthreads();

    // staging geometry: thread owns column fB, K-pairs kp = kp0 + 2*i
    const int fB  = tid & 127;
    const int kp0 = tid >> 7;                    // 0 or 1
    const float* gwp = gate_ws + (size_t)e * D_DIM * F_DIM + n0 + fB;
    const float* uwp = up_ws   + (size_t)e * D_DIM * F_DIM + n0 + fB;

    f32x8 accG[4], accU[4];
    #pragma unroll
    for (int t = 0; t < 4; ++t) { accG[t] = (f32x8){}; accU[t] = (f32x8){}; }

    const int ar = tid >> 2, ac = tid & 3;       // 64 rows x 4 chunks of 16B
    const uint32_t ldsA_off =
        (uint32_t)(uintptr_t)(&ldsA[0]) + (uint32_t)(ar * 32 + ac * 8) * 2;
    const __bf16* arow = xbf + (size_t)sRow[ar] * D_DIM + ac * 8;

    for (int kc = 0; kc < D_DIM / 32; ++kc) {
        const int kb = kc * 32;
        async_copy_b128(ldsA_off, (uint64_t)(uintptr_t)(arow + kb));

        // load all B values first (coalesced b32, independent regs), then pack+store
        float g0[8], g1[8], u0[8], u1[8];
        #pragma unroll
        for (int i = 0; i < 8; ++i) {
            int kp = kp0 + 2 * i;
            size_t o = (size_t)(kb + 2 * kp) * F_DIM;
            g0[i] = gwp[o]; g1[i] = gwp[o + F_DIM];
            u0[i] = uwp[o]; u1[i] = uwp[o + F_DIM];
        }
        #pragma unroll
        for (int i = 0; i < 8; ++i) {
            int kp = kp0 + 2 * i;
            *(uint32_t*)&ldsG[fB * 40 + 2 * kp] = pack2bf16(g0[i], g1[i]);
            *(uint32_t*)&ldsU[fB * 40 + 2 * kp] = pack2bf16(u0[i], u1[i]);
        }
        wait_async();
        __syncthreads();

        FragB16 a;
        {
            const __bf16* ap = &ldsA[(wm * 16 + (lane & 15)) * 32 + (lane >> 4) * 8];
            a.q[0] = *(const u32x4*)ap;
            a.q[1] = *(const u32x4*)(ap + 16);
        }
        const int k0 = (lane >> 4) * 8;
        #pragma unroll
        for (int t = 0; t < 4; ++t) {
            int floc = wn * 64 + t * 16 + (lane & 15);
            FragB16 b;
            const __bf16* bg = &ldsG[floc * 40 + k0];
            b.q[0] = *(const u32x4*)bg; b.q[1] = *(const u32x4*)(bg + 16);
            accG[t] = __builtin_amdgcn_wmma_f32_16x16x32_bf16(
                false, a.v, false, b.v, (short)0, accG[t], false, false);
            const __bf16* bu = &ldsU[floc * 40 + k0];
            b.q[0] = *(const u32x4*)bu; b.q[1] = *(const u32x4*)(bu + 16);
            accU[t] = __builtin_amdgcn_wmma_f32_16x16x32_bf16(
                false, a.v, false, b.v, (short)0, accU[t], false, false);
        }
        __syncthreads();
    }

    const int mb = wm * 16 + (lane >> 4) * 8;    // C layout: m = 8*(lane/16)+r
    #pragma unroll
    for (int t = 0; t < 4; ++t) {
        int f = n0 + wn * 64 + t * 16 + (lane & 15);
        #pragma unroll
        for (int r = 0; r < 8; ++r) {
            int mloc = mb + r;
            if (m0 + mloc < cnt) {
                float g = accG[t][r], u = accU[t][r];
                float sg = g / (1.0f + __expf(-g));
                hbuf[(size_t)sEnt[mloc] * F_DIM + f] = (__bf16)(sg * u);
            }
        }
    }
}

// ---------------- down GEMM + weighted scatter-add --------------------------
__global__ __launch_bounds__(256) void down_kernel(
    const __bf16* __restrict__ hbuf, const float* __restrict__ down_ws,
    const int* __restrict__ counts, const int* __restrict__ lists,
    const float* __restrict__ wts, float* __restrict__ out) {
    const int e = blockIdx.z;
    const int cnt = counts[e];
    const int m0 = blockIdx.y * MT;
    if (m0 >= cnt) return;
    const int n0 = blockIdx.x * NT;
    const int tid = threadIdx.x;
    const int lane = tid & 31, wave = tid >> 5;
    const int wm = wave & 3, wn = wave >> 2;

    __shared__ __bf16 ldsA[MT * 32];
    __shared__ __bf16 ldsB[NT * 40];
    __shared__ int   sEnt[MT];
    __shared__ int   sTok[MT];
    __shared__ float sW[MT];

    if (tid < MT) {
        int idx = m0 + tid; if (idx >= cnt) idx = cnt - 1;
        int ent = lists[e * NTOK + idx];
        sEnt[tid] = ent; sTok[tid] = ent >> 1; sW[tid] = wts[e * NTOK + idx];
    }
    __syncthreads();

    const int fB  = tid & 127;
    const int kp0 = tid >> 7;
    const float* dwp = down_ws + (size_t)e * F_DIM * D_DIM + n0 + fB;

    f32x8 acc[4];
    #pragma unroll
    for (int t = 0; t < 4; ++t) acc[t] = (f32x8){};

    const int ar = tid >> 2, ac = tid & 3;
    const uint32_t ldsA_off =
        (uint32_t)(uintptr_t)(&ldsA[0]) + (uint32_t)(ar * 32 + ac * 8) * 2;
    const __bf16* arow = hbuf + (size_t)sEnt[ar] * F_DIM + ac * 8;

    for (int kc = 0; kc < F_DIM / 32; ++kc) {        // 88 K-chunks
        const int kb = kc * 32;
        async_copy_b128(ldsA_off, (uint64_t)(uintptr_t)(arow + kb));

        float d0[8], d1[8];
        #pragma unroll
        for (int i = 0; i < 8; ++i) {
            int kp = kp0 + 2 * i;
            size_t o = (size_t)(kb + 2 * kp) * D_DIM;
            d0[i] = dwp[o]; d1[i] = dwp[o + D_DIM];
        }
        #pragma unroll
        for (int i = 0; i < 8; ++i) {
            int kp = kp0 + 2 * i;
            *(uint32_t*)&ldsB[fB * 40 + 2 * kp] = pack2bf16(d0[i], d1[i]);
        }
        wait_async();
        __syncthreads();

        FragB16 a;
        {
            const __bf16* ap = &ldsA[(wm * 16 + (lane & 15)) * 32 + (lane >> 4) * 8];
            a.q[0] = *(const u32x4*)ap;
            a.q[1] = *(const u32x4*)(ap + 16);
        }
        const int k0 = (lane >> 4) * 8;
        #pragma unroll
        for (int t = 0; t < 4; ++t) {
            int floc = wn * 64 + t * 16 + (lane & 15);
            FragB16 b;
            const __bf16* bp = &ldsB[floc * 40 + k0];
            b.q[0] = *(const u32x4*)bp; b.q[1] = *(const u32x4*)(bp + 16);
            acc[t] = __builtin_amdgcn_wmma_f32_16x16x32_bf16(
                false, a.v, false, b.v, (short)0, acc[t], false, false);
        }
        __syncthreads();
    }

    const int mb = wm * 16 + (lane >> 4) * 8;
    #pragma unroll
    for (int t = 0; t < 4; ++t) {
        int dcol = n0 + wn * 64 + t * 16 + (lane & 15);
        #pragma unroll
        for (int r = 0; r < 8; ++r) {
            int mloc = mb + r;
            if (m0 + mloc < cnt)
                unsafeAtomicAdd(&out[(size_t)sTok[mloc] * D_DIM + dcol],
                                sW[mloc] * acc[t][r]);
        }
    }
}

// ---------------- host launch ----------------------------------------------
extern "C" void kernel_launch(void* const* d_in, const int* in_sizes, int n_in,
                              void* d_out, int out_size, void* d_ws, size_t ws_size,
                              hipStream_t stream) {
    (void)in_sizes; (void)n_in; (void)out_size; (void)ws_size;
    const float* x       = (const float*)d_in[0];   // [2,2048,1024]
    const float* gate_w  = (const float*)d_in[1];   // [8,1024]
    const float* gate_ws = (const float*)d_in[2];   // [8,1024,2816]
    const float* up_ws   = (const float*)d_in[3];   // [8,1024,2816]
    const float* down_ws = (const float*)d_in[4];   // [8,2816,1024]
    float* out    = (float*)d_out;                  // [4096*1024]
    float* logits = out + (size_t)NTOK * D_DIM;     // [4096*8]

    char* ws = (char*)d_ws;
    int*    counts = (int*)ws;                                       // 32 B
    int*    lists  = (int*)(ws + 256);                               // 8*4096*4
    float*  wts    = (float*)(ws + 256 + (size_t)NEXP * NTOK * 4);   // 8*4096*4
    __bf16* xbf    = (__bf16*)(ws + 256 + 2ull * NEXP * NTOK * 4);   // 8.4 MB
    __bf16* hbuf   = (__bf16*)(ws + 256 + 2ull * NEXP * NTOK * 4
                               + (size_t)NTOK * D_DIM * 2);          // 46.1 MB

    init_kernel  <<<4096, 256, 0, stream>>>(out, counts);
    cvt_kernel   <<<4096, 256, 0, stream>>>(x, xbf);
    router_kernel<<<NTOK / 8, 256, 0, stream>>>(x, gate_w, logits, counts, lists, wts);
    gateup_kernel<<<dim3(F_DIM / NT, NTOK / MT, NEXP), 256, 0, stream>>>(
        xbf, gate_ws, up_ws, counts, lists, hbuf);
    down_kernel  <<<dim3(D_DIM / NT, NTOK / MT, NEXP), 256, 0, stream>>>(
        hbuf, down_ws, counts, lists, wts, out);
}
